// Gpt2Attention_19774029431582
// MI455X (gfx1250) — compile-verified, matrix operand order checked by reading, not run on hardware
//
#include <hip/hip_runtime.h>

typedef __attribute__((ext_vector_type(16))) _Float16 v16h;
typedef __attribute__((ext_vector_type(8)))  _Float16 v8h;
typedef __attribute__((ext_vector_type(8)))  float    v8f;

#define B_   2
#define S_   2048
#define D_   1024
#define H_   16
#define K_   64
#define BS_  (B_*S_)      // 4096
#define NQKV (3*H_*K_)    // 3072

// Build a WMMA A/B fragment (v16h) from two contiguous 8-half segments.
// Fragment layout (ISA 7.12.2): elems 0..7  = src row, K = half*8 + 0..7
//                               elems 8..15 = src row, K = 16 + half*8 + 0..7
static __device__ __forceinline__ v16h frag16(const _Float16* p0,
                                              const _Float16* p1) {
  v8h lo = *(const v8h*)p0;
  v8h hi = *(const v8h*)p1;
  return __builtin_shufflevector(lo, hi, 0, 1, 2, 3, 4, 5, 6, 7,
                                 8, 9, 10, 11, 12, 13, 14, 15);
}

// ---------------------------------------------------------------------------
// One-shot f32 -> f16 convert (for x). Fully coalesced both sides.
// ---------------------------------------------------------------------------
__global__ __launch_bounds__(256) void cvt_f16_kernel(
    const float* __restrict__ src, _Float16* __restrict__ dst) {
  const size_t idx = (size_t)blockIdx.x * 256 + threadIdx.x;
  dst[idx] = (_Float16)src[idx];
}

// ---------------------------------------------------------------------------
// Transpose + f16-convert weights: wt[n][k] = (f16) w[k][n].
// 2D grid: blockIdx.y = k row (no integer division anywhere).
// One-shot; ~17MB traffic total, negligible at 23.3 TB/s.
// ---------------------------------------------------------------------------
__global__ __launch_bounds__(256) void transpose_w_kernel(
    const float* __restrict__ w, _Float16* __restrict__ wt, int kdim, int n) {
  const int nn = blockIdx.x * 256 + threadIdx.x;   // coalesced read along n
  const int kk = blockIdx.y;
  wt[(size_t)nn * kdim + kk] = (_Float16)w[(size_t)kk * n + nn];
}

// ---------------------------------------------------------------------------
// Kernel A: QKV projection.  xh(4096x1024 f16) * Wt(3072x1024 f16, pre-T)
// -> Q/K f16 [b][h][s][k] (Q pre-scaled 1/8), V f16 transposed [b][h][k][s].
// One wave = one 16x16 tile; 32x v_wmma_f32_16x16x32_f16; all operand
// traffic is contiguous b128 loads.
// ---------------------------------------------------------------------------
__global__ __launch_bounds__(256) void qkv_kernel(
    const _Float16* __restrict__ xh, const _Float16* __restrict__ wt,
    const float* __restrict__ bias,
    _Float16* __restrict__ Qh, _Float16* __restrict__ Kh,
    _Float16* __restrict__ VT)
{
  const int lane  = threadIdx.x & 31;
  const int wave  = threadIdx.x >> 5;
  const int tile  = blockIdx.x * 8 + wave;
  const int mtile = tile / (NQKV / 16);   // 0..255
  const int ntile = tile % (NQKV / 16);   // 0..191
  const int half  = lane >> 4;
  const int l16   = lane & 15;
  const int row   = mtile * 16 + l16;     // A row (global token)
  const int ncol  = ntile * 16 + l16;     // B column

  const _Float16* arow = xh + (size_t)row * D_;
  const _Float16* brow = wt + (size_t)ncol * D_;

  v8f c = {};
  for (int kb = 0; kb < D_; kb += 32) {
    v16h a = frag16(arow + kb + half * 8, arow + kb + 16 + half * 8);
    v16h b = frag16(brow + kb + half * 8, brow + kb + 16 + half * 8);
    c = __builtin_amdgcn_wmma_f32_16x16x32_f16(false, a, false, b, (short)0, c,
                                               false, false);
  }

  const float bn = bias[ncol];
  const int t  = ncol / (H_ * K_);
  const int hh = (ncol % (H_ * K_)) / K_;
  const int kk = ncol % K_;
#pragma unroll
  for (int r = 0; r < 8; ++r) {
    const int grow = mtile * 16 + r + 8 * half;
    const int bb = grow >> 11;            // /S_
    const int ss = grow & (S_ - 1);
    const float v = c[r] + bn;
    if (t == 0)
      Qh[((size_t)(bb * H_ + hh) * S_ + ss) * K_ + kk] = (_Float16)(v * 0.125f);
    else if (t == 1)
      Kh[((size_t)(bb * H_ + hh) * S_ + ss) * K_ + kk] = (_Float16)v;
    else  // V stored transposed: [b][h][dim][s]
      VT[((size_t)(bb * H_ + hh) * K_ + kk) * S_ + ss] = (_Float16)v;
  }
}

// ---------------------------------------------------------------------------
// Kernel B: causal flash attention. One wave owns a 16-row query tile of one
// (b,h); iterates 32-key blocks with online softmax; scores stay in VGPRs.
// All fragments are 2x b128 loads; V comes from the transposed layout.
// ---------------------------------------------------------------------------
__global__ __launch_bounds__(256) void attn_kernel(
    const _Float16* __restrict__ Qh, const _Float16* __restrict__ Kh,
    const _Float16* __restrict__ VT, _Float16* __restrict__ AO)
{
  __shared__ _Float16 lds[8][16 * 32];   // per-wave P staging (1KB)
  const int lane = threadIdx.x & 31;
  const int wave = threadIdx.x >> 5;
  _Float16* P = &lds[wave][0];

  const int tile = blockIdx.x * 8 + wave;     // 0..4095
  const int qt   = tile & 127;
  const int h    = (tile >> 7) & 15;
  const int b    = tile >> 11;
  const int half = lane >> 4;
  const int l16  = lane & 15;
  const int qbase = qt * 16;

  const _Float16* Qb  = Qh + (size_t)(b * H_ + h) * S_ * K_;
  const _Float16* Kb  = Kh + (size_t)(b * H_ + h) * S_ * K_;
  const _Float16* VTb = VT + (size_t)(b * H_ + h) * K_ * S_;  // [dim][s]

  // Q tile (16x64) -> two resident A fragments.
  const _Float16* qrow = Qb + (size_t)(qbase + l16) * K_;
  v16h aq0 = frag16(qrow + half * 8, qrow + 16 + half * 8);
  v16h aq1 = frag16(qrow + 32 + half * 8, qrow + 48 + half * 8);

  v8f o[4] = {v8f{}, v8f{}, v8f{}, v8f{}};   // 16x64 f32 accumulator
  float m8[8], l8[8];
#pragma unroll
  for (int r = 0; r < 8; ++r) { m8[r] = -1e30f; l8[r] = 0.0f; }

  const int nkb = (qbase + 47) / 32;          // causal: keys 0..qbase+15
  for (int jb = 0; jb < nkb; ++jb) {
    const int kb2 = jb * 32;

    if (jb + 1 < nkb) {  // gfx1250 global_prefetch of next K/V block
      __builtin_prefetch(Kb + (size_t)(kb2 + 32 + lane) * K_, 0, 0);
      __builtin_prefetch(VTb + (size_t)lane * S_ + kb2 + 32, 0, 0);
    }

    // ---- scores: two 16x16 tiles (keys kb2..+15, +16..+31) ----
    v8f s0 = {}, s1 = {};
    {
      const _Float16* krow = Kb + (size_t)(kb2 + l16) * K_;
      v16h bk0 = frag16(krow + half * 8, krow + 16 + half * 8);
      v16h bk1 = frag16(krow + 32 + half * 8, krow + 48 + half * 8);
      s0 = __builtin_amdgcn_wmma_f32_16x16x32_f16(false, aq0, false, bk0,
                                                  (short)0, s0, false, false);
      s0 = __builtin_amdgcn_wmma_f32_16x16x32_f16(false, aq1, false, bk1,
                                                  (short)0, s0, false, false);
    }
    {
      const _Float16* krow = Kb + (size_t)(kb2 + 16 + l16) * K_;
      v16h bk0 = frag16(krow + half * 8, krow + 16 + half * 8);
      v16h bk1 = frag16(krow + 32 + half * 8, krow + 48 + half * 8);
      s1 = __builtin_amdgcn_wmma_f32_16x16x32_f16(false, aq0, false, bk0,
                                                  (short)0, s1, false, false);
      s1 = __builtin_amdgcn_wmma_f32_16x16x32_f16(false, aq1, false, bk1,
                                                  (short)0, s1, false, false);
    }

    // ---- causal mask + online softmax (row r lives in a 16-lane half) ----
    const int key0 = kb2 + l16;
    const int key1 = kb2 + 16 + l16;
#pragma unroll
    for (int r = 0; r < 8; ++r) {
      const int qrw = qbase + r + 8 * half;
      if (key0 > qrw) s0[r] = -1e9f;
      if (key1 > qrw) s1[r] = -1e9f;

      float rm = fmaxf(s0[r], s1[r]);
      rm = fmaxf(rm, __shfl_xor(rm, 1, 32));
      rm = fmaxf(rm, __shfl_xor(rm, 2, 32));
      rm = fmaxf(rm, __shfl_xor(rm, 4, 32));
      rm = fmaxf(rm, __shfl_xor(rm, 8, 32));

      const float mnew  = fmaxf(m8[r], rm);
      const float scale = __expf(m8[r] - mnew);
      const float p0 = __expf(s0[r] - mnew);
      const float p1 = __expf(s1[r] - mnew);
      float rs = p0 + p1;
      rs += __shfl_xor(rs, 1, 32);
      rs += __shfl_xor(rs, 2, 32);
      rs += __shfl_xor(rs, 4, 32);
      rs += __shfl_xor(rs, 8, 32);

      l8[r] = l8[r] * scale + rs;
      m8[r] = mnew;
#pragma unroll
      for (int cc = 0; cc < 4; ++cc) o[cc][r] *= scale;

      // stage P row-major [m][k] for the A-layout reload
      P[(r + 8 * half) * 32 + l16]      = (_Float16)p0;
      P[(r + 8 * half) * 32 + 16 + l16] = (_Float16)p1;
    }

    // same-wave LDS is in-order; fence compiler + drain DS returns
    asm volatile("s_wait_dscnt 0x0" ::: "memory");

    // P (16x32) A fragment: two contiguous ds_load_b128 per lane
    const _Float16* prow = P + l16 * 32;
    v16h ap = frag16(prow + half * 8, prow + 16 + half * 8);

    // ---- O += P(16x32) x V(32x64), V from transposed layout ----
#pragma unroll
    for (int cc = 0; cc < 4; ++cc) {
      const _Float16* vrow = VTb + (size_t)(cc * 16 + l16) * S_ + kb2;
      v16h bv = frag16(vrow + half * 8, vrow + 16 + half * 8);
      o[cc] = __builtin_amdgcn_wmma_f32_16x16x32_f16(false, ap, false, bv,
                                                     (short)0, o[cc], false, false);
    }
  }

  // ---- normalize, store attn_out f16 as (4096 x 1024) row-major ----
#pragma unroll
  for (int r = 0; r < 8; ++r) {
    const float inv = 1.0f / l8[r];
    const int srow = qbase + r + 8 * half;
#pragma unroll
    for (int cc = 0; cc < 4; ++cc) {
      const int d = h * K_ + cc * 16 + l16;
      AO[((size_t)(b * S_ + srow)) * (H_ * K_) + d] = (_Float16)(o[cc][r] * inv);
    }
  }
}

// ---------------------------------------------------------------------------
// Kernel C: output projection. AO(4096x1024 f16) * Wt_proj(1024x1024 f16,
// pre-transposed) + b_proj -> out f32.
// ---------------------------------------------------------------------------
__global__ __launch_bounds__(256) void proj_kernel(
    const _Float16* __restrict__ AO, const _Float16* __restrict__ wt,
    const float* __restrict__ bias, float* __restrict__ out)
{
  const int lane  = threadIdx.x & 31;
  const int wave  = threadIdx.x >> 5;
  const int tile  = blockIdx.x * 8 + wave;
  const int mtile = tile / (D_ / 16);     // 0..255
  const int ntile = tile % (D_ / 16);     // 0..63
  const int half  = lane >> 4;
  const int l16   = lane & 15;
  const int row   = mtile * 16 + l16;
  const int ncol  = ntile * 16 + l16;

  const _Float16* arow = AO + (size_t)row * D_;
  const _Float16* brow = wt + (size_t)ncol * D_;

  v8f c = {};
  for (int kb = 0; kb < D_; kb += 32) {
    v16h a = frag16(arow + kb + half * 8, arow + kb + 16 + half * 8);
    v16h b = frag16(brow + kb + half * 8, brow + kb + 16 + half * 8);
    c = __builtin_amdgcn_wmma_f32_16x16x32_f16(false, a, false, b, (short)0, c,
                                               false, false);
  }
  const float bn = bias[ncol];
#pragma unroll
  for (int r = 0; r < 8; ++r) {
    const int grow = mtile * 16 + r + 8 * half;
    out[(size_t)grow * D_ + ncol] = c[r] + bn;
  }
}

// ---------------------------------------------------------------------------
extern "C" void kernel_launch(void* const* d_in, const int* in_sizes, int n_in,
                              void* d_out, int out_size, void* d_ws, size_t ws_size,
                              hipStream_t stream) {
  (void)in_sizes; (void)n_in; (void)out_size; (void)ws_size;
  const float* x      = (const float*)d_in[0];
  const float* mask   = (const float*)d_in[1];   // causal mask applied analytically
  const float* W_qkv  = (const float*)d_in[2];
  const float* b_qkv  = (const float*)d_in[3];
  const float* W_proj = (const float*)d_in[4];
  const float* b_proj = (const float*)d_in[5];
  (void)mask;                                    // d_in[6] = layer_idx, unused
  float* out = (float*)d_out;

  char* ws = (char*)d_ws;
  const size_t qsz = (size_t)B_ * H_ * S_ * K_ * sizeof(_Float16);  // 8 MB
  _Float16* Qh  = (_Float16*)(ws);
  _Float16* Kh  = (_Float16*)(ws + qsz);
  _Float16* VT  = (_Float16*)(ws + 2 * qsz);
  _Float16* AO  = (_Float16*)(ws + 3 * qsz);
  _Float16* Xh  = (_Float16*)(ws + 4 * qsz);                          // 8 MB
  _Float16* Wtq = (_Float16*)(ws + 5 * qsz);                          // 6 MB
  _Float16* Wtp = (_Float16*)(ws + 5 * qsz + (size_t)D_ * NQKV * sizeof(_Float16));

  // One-shot input/weight conversion into WMMA-friendly f16 layouts.
  cvt_f16_kernel<<<(BS_ * D_) / 256, 256, 0, stream>>>(x, Xh);
  transpose_w_kernel<<<dim3(NQKV / 256, D_), 256, 0, stream>>>(W_qkv, Wtq, D_, NQKV);
  transpose_w_kernel<<<dim3(D_ / 256, D_), 256, 0, stream>>>(W_proj, Wtp, D_, D_);

  // (4096/16)*(3072/16) = 49152 wave-tiles, 8 waves/block
  qkv_kernel<<<(BS_ / 16) * (NQKV / 16) / 8, 256, 0, stream>>>(
      Xh, Wtq, b_qkv, Qh, Kh, VT);
  // B*H*(S/16) = 4096 wave-tiles
  attn_kernel<<<(B_ * H_ * (S_ / 16)) / 8, 256, 0, stream>>>(Qh, Kh, VT, AO);
  // (4096/16)*(1024/16) = 16384 wave-tiles
  proj_kernel<<<(BS_ / 16) * (D_ / 16) / 8, 256, 0, stream>>>(
      AO, Wtp, b_proj, out);
}
